// TransformersLSTM_34840774705785
// MI455X (gfx1250) — matrix operand order repaired and checked by reading, not compiled, not run on hardware
//
#include <hip/hip_runtime.h>

// ---------------------------------------------------------------------------
// Attention-LSTM seq2seq, MI455X (gfx1250, wave32).
// Bandwidth-bound batch-1 recurrence: weights converted to bf16 once per call
// (~38MB, L2-resident on the 192MB L2), matvecs done with
// v_wmma_f32_16x16x32_bf16 (16 rows per wave, activation broadcast over N),
// f32 state/accumulators. Attention reduced to a single-row dot product since
// the reference only consumes energy[i].
// ---------------------------------------------------------------------------

typedef __bf16 bf16_t;
typedef __attribute__((ext_vector_type(16))) __bf16 v16bf;
typedef __attribute__((ext_vector_type(8)))  __bf16 v8bf;
typedef __attribute__((ext_vector_type(8)))  float  v8f;

#define T_STEPS 512
#define I_DIM   512
#define H_DIM   1024
#define G_DIM   4096   // 4*H
#define L_DIM   512

__device__ __forceinline__ bf16_t f2bf(float f) {
  unsigned u = __builtin_bit_cast(unsigned, f);
  unsigned r = u + 0x7FFFu + ((u >> 16) & 1u);   // round to nearest even
  unsigned short h = (unsigned short)(r >> 16);
  return __builtin_bit_cast(bf16_t, h);
}

__device__ __forceinline__ float sigmoidf_(float x) { return 1.0f / (1.0f + expf(-x)); }

// --------------------------- pre-pass kernels ------------------------------

__global__ void cvt_f32_bf16(const float* __restrict__ src, bf16_t* __restrict__ dst, int n) {
  int i = blockIdx.x * blockDim.x + threadIdx.x;
  if (i < n) dst[i] = f2bf(src[i]);
}

__global__ void init_state(float* __restrict__ eh, float* __restrict__ ec,
                           float* __restrict__ dh, float* __restrict__ dc,
                           float* __restrict__ aw,
                           bf16_t* __restrict__ eh_h, bf16_t* __restrict__ dh_h,
                           bf16_t* __restrict__ din_h) {
  int i = threadIdx.x;                 // 1024 threads
  eh[i] = 0.f; ec[i] = 0.f; dh[i] = 0.f; dc[i] = 0.f;
  if (i < L_DIM) aw[i] = 0.f;
  bf16_t z = f2bf(0.f);
  eh_h[i] = z; dh_h[i] = z;
  din_h[i] = z; din_h[H_DIM + i] = z;
}

// --------------------------- WMMA matvec -----------------------------------
// Wave computes y[row0 .. row0+15] += W[rows, :] @ x using 16x16x32 bf16 WMMA.
// A layout (16-bit A 16x32, ISA 7.12.2): lane L holds M = L%16,
//   K = klo..klo+7 (VGPR0-3) and 16+klo..16+klo+7 (VGPR4-7), klo = L<16?0:8.
// B layout (32x16): lane L holds K = kB..kB+7 and 16+kB..16+kB+7, kB = L<16?0:8;
//   all N columns carry the same broadcast vector x.
__device__ __forceinline__ void wave_matvec_acc(const bf16_t* __restrict__ W, int ld,
                                                const bf16_t* __restrict__ x, int K,
                                                int row0, int lane, v8f& c) {
  const int m    = row0 + (lane & 15);
  const int klo  = (lane < 16) ? 0 : 8;
  const int kB   = (lane < 16) ? 0 : 8;
  const bf16_t* wrow = W + (size_t)m * (size_t)ld;
  for (int kb = 0; kb < K; kb += 32) {
    v8bf a_lo = *(const v8bf*)(wrow + kb + klo);
    v8bf a_hi = *(const v8bf*)(wrow + kb + 16 + klo);
    v8bf b_lo = *(const v8bf*)(x + kb + kB);
    v8bf b_hi = *(const v8bf*)(x + kb + 16 + kB);
    v16bf a = __builtin_shufflevector(a_lo, a_hi, 0,1,2,3,4,5,6,7,8,9,10,11,12,13,14,15);
    v16bf b = __builtin_shufflevector(b_lo, b_hi, 0,1,2,3,4,5,6,7,8,9,10,11,12,13,14,15);
    c = __builtin_amdgcn_wmma_f32_16x16x32_bf16(false, a, false, b, (short)0, c, false, false);
  }
}

// gates = W1 @ x1 + W2 @ x2   (biases added in the cell kernels)
// grid: ROWS/128 blocks of 256 threads (8 waves), 16 rows per wave.
__global__ void __launch_bounds__(256)
gates_kernel(const bf16_t* __restrict__ W1, int ld1, const bf16_t* __restrict__ x1, int K1,
             const bf16_t* __restrict__ W2, int ld2, const bf16_t* __restrict__ x2, int K2,
             float* __restrict__ gates) {
  const int lane  = (int)(threadIdx.x & 31u);
  const int gwave = (int)(blockIdx.x * (blockDim.x >> 5) + (threadIdx.x >> 5));
  const int row0  = gwave * 16;
  v8f c = {0.f, 0.f, 0.f, 0.f, 0.f, 0.f, 0.f, 0.f};
  wave_matvec_acc(W1, ld1, x1, K1, row0, lane, c);
  wave_matvec_acc(W2, ld2, x2, K2, row0, lane, c);
  // D layout: VGPR v at lane L -> row = row0 + v + (L<16 ? 0 : 8); all N equal.
  if ((lane & 15) == 0) {
    const int rbase = row0 + ((lane >> 4) << 3);
    #pragma unroll
    for (int v = 0; v < 8; ++v) gates[rbase + v] = c[v];
  }
}

// --------------------------- elementwise cells -----------------------------

__global__ void __launch_bounds__(1024)
lstm_cell_enc(const float* __restrict__ gates, const float* __restrict__ bih,
              const float* __restrict__ bhh,
              float* __restrict__ h, float* __restrict__ cst, bf16_t* __restrict__ h_bf) {
  const int j = (int)threadIdx.x;
  float gi = gates[j]            + bih[j]            + bhh[j];
  float gf = gates[H_DIM + j]    + bih[H_DIM + j]    + bhh[H_DIM + j];
  float gg = gates[2*H_DIM + j]  + bih[2*H_DIM + j]  + bhh[2*H_DIM + j];
  float go = gates[3*H_DIM + j]  + bih[3*H_DIM + j]  + bhh[3*H_DIM + j];
  float c  = sigmoidf_(gf) * cst[j] + sigmoidf_(gi) * tanhf(gg);
  float hn = sigmoidf_(go) * tanhf(c);
  cst[j] = c; h[j] = hn; h_bf[j] = f2bf(hn);
}

__global__ void __launch_bounds__(1024)
lstm_cell_dec(const float* __restrict__ gates, const float* __restrict__ bih,
              const float* __restrict__ bhh,
              float* __restrict__ h, float* __restrict__ cst,
              bf16_t* __restrict__ h_bf, bf16_t* __restrict__ din_hi,
              const float* __restrict__ W_out, const float* __restrict__ b_out,
              float* __restrict__ out_t) {
  __shared__ float red[1024];
  const int j = (int)threadIdx.x;
  float gi = gates[j]            + bih[j]            + bhh[j];
  float gf = gates[H_DIM + j]    + bih[H_DIM + j]    + bhh[H_DIM + j];
  float gg = gates[2*H_DIM + j]  + bih[2*H_DIM + j]  + bhh[2*H_DIM + j];
  float go = gates[3*H_DIM + j]  + bih[3*H_DIM + j]  + bhh[3*H_DIM + j];
  float c  = sigmoidf_(gf) * cst[j] + sigmoidf_(gi) * tanhf(gg);
  float hn = sigmoidf_(go) * tanhf(c);
  cst[j] = c; h[j] = hn;
  bf16_t hb = f2bf(hn);
  h_bf[j] = hb; din_hi[j] = hb;
  // out_t = W_out @ dh + b_out  (O = 1)
  red[j] = W_out[j] * hn;
  __syncthreads();
  for (int s = 512; s > 0; s >>= 1) { if (j < s) red[j] += red[j + s]; __syncthreads(); }
  if (j == 0) *out_t = red[0] + b_out[0];
}

// --------------------------- attention -------------------------------------
// Only energy[t] is consumed by the reference, so compute just row t of
// W_attn, then re-softmax the persistent aw vector and build context.
__global__ void __launch_bounds__(1024)
attn_kernel(const float* __restrict__ W_attn, const float* __restrict__ b_attn,
            const float* __restrict__ eh, const float* __restrict__ dh,
            float* __restrict__ aw, bf16_t* __restrict__ din_lo, int t) {
  __shared__ float red[1024];
  __shared__ float sE[L_DIM];
  const int j = (int)threadIdx.x;
  // energy[t] = W_attn[t,:] . concat(eh, dh) + b_attn[t]
  const float* wrow = W_attn + (size_t)t * (size_t)(2 * H_DIM);
  red[j] = wrow[j] * eh[j] + wrow[H_DIM + j] * dh[j];
  __syncthreads();
  for (int s = 512; s > 0; s >>= 1) { if (j < s) red[j] += red[j + s]; __syncthreads(); }
  float energy_t = red[0] + b_attn[t];
  __syncthreads();
  // aw[t] = energy_t, then softmax(aw)
  if (j < L_DIM) sE[j] = (j == t) ? energy_t : aw[j];
  red[j] = (j < L_DIM) ? ((j == t) ? energy_t : aw[j]) : -3.402823466e38f;
  __syncthreads();
  for (int s = 512; s > 0; s >>= 1) { if (j < s) red[j] = fmaxf(red[j], red[j + s]); __syncthreads(); }
  float mx = red[0];
  float e = (j < L_DIM) ? expf(sE[j] - mx) : 0.f;
  __syncthreads();                       // done reading red[0] / sE
  if (j < L_DIM) sE[j] = e;
  red[j] = e;
  __syncthreads();
  for (int s = 512; s > 0; s >>= 1) { if (j < s) red[j] += red[j + s]; __syncthreads(); }
  float inv = 1.0f / red[0];
  if (j < L_DIM) aw[j] = sE[j] * inv;
  float s_attn = sE[t] * inv;            // aw[t] after softmax
  din_lo[j] = f2bf(s_attn * eh[j]);      // context -> decoder input (low half)
}

// --------------------------- host side -------------------------------------

extern "C" void kernel_launch(void* const* d_in, const int* in_sizes, int n_in,
                              void* d_out, int out_size, void* d_ws, size_t ws_size,
                              hipStream_t stream) {
  (void)in_sizes; (void)n_in; (void)out_size; (void)ws_size;

  const float* input_seq = (const float*)d_in[0];
  const float* W_ih_e = (const float*)d_in[2];
  const float* W_hh_e = (const float*)d_in[3];
  const float* b_ih_e = (const float*)d_in[4];
  const float* b_hh_e = (const float*)d_in[5];
  const float* W_attn = (const float*)d_in[6];
  const float* b_attn = (const float*)d_in[7];
  const float* W_ih_d = (const float*)d_in[8];
  const float* W_hh_d = (const float*)d_in[9];
  const float* b_ih_d = (const float*)d_in[10];
  const float* b_hh_d = (const float*)d_in[11];
  const float* W_out  = (const float*)d_in[12];
  const float* b_out  = (const float*)d_in[13];
  float* out = (float*)d_out;

  // workspace carve-out (256B-aligned slabs)
  char* ws = (char*)d_ws;
  size_t off = 0;
  auto alloc = [&](size_t bytes) -> char* {
    char* p = ws + off;
    off = (off + bytes + 255) & ~(size_t)255;
    return p;
  };
  bf16_t* Wihe_h = (bf16_t*)alloc((size_t)G_DIM * I_DIM * 2);
  bf16_t* Whhe_h = (bf16_t*)alloc((size_t)G_DIM * H_DIM * 2);
  bf16_t* Wihd_h = (bf16_t*)alloc((size_t)G_DIM * 2 * H_DIM * 2);
  bf16_t* Whhd_h = (bf16_t*)alloc((size_t)G_DIM * H_DIM * 2);
  bf16_t* x_h    = (bf16_t*)alloc((size_t)T_STEPS * I_DIM * 2);
  bf16_t* eh_h   = (bf16_t*)alloc((size_t)H_DIM * 2);
  bf16_t* dh_h   = (bf16_t*)alloc((size_t)H_DIM * 2);
  bf16_t* din_h  = (bf16_t*)alloc((size_t)2 * H_DIM * 2);
  float*  eh     = (float*)alloc((size_t)H_DIM * 4);
  float*  ec     = (float*)alloc((size_t)H_DIM * 4);
  float*  dh     = (float*)alloc((size_t)H_DIM * 4);
  float*  dc     = (float*)alloc((size_t)H_DIM * 4);
  float*  aw     = (float*)alloc((size_t)L_DIM * 4);
  float*  gates_e = (float*)alloc((size_t)G_DIM * 4);
  float*  gates_d = (float*)alloc((size_t)G_DIM * 4);

  // one-time (per call) bf16 conversion of the streamed operands
  auto cvt = [&](const float* src, bf16_t* dst, int n) {
    int blocks = (n + 255) / 256;
    cvt_f32_bf16<<<blocks, 256, 0, stream>>>(src, dst, n);
  };
  cvt(W_ih_e, Wihe_h, G_DIM * I_DIM);
  cvt(W_hh_e, Whhe_h, G_DIM * H_DIM);
  cvt(W_ih_d, Wihd_h, G_DIM * 2 * H_DIM);
  cvt(W_hh_d, Whhd_h, G_DIM * H_DIM);
  cvt(input_seq, x_h, T_STEPS * I_DIM);
  init_state<<<1, 1024, 0, stream>>>(eh, ec, dh, dc, aw, eh_h, dh_h, din_h);

  // 512 sequential timesteps; 4096 gate rows -> 256 waves -> 32 blocks x 256
  for (int t = 0; t < T_STEPS; ++t) {
    gates_kernel<<<32, 256, 0, stream>>>(Wihe_h, I_DIM, x_h + (size_t)t * I_DIM, I_DIM,
                                         Whhe_h, H_DIM, eh_h, H_DIM, gates_e);
    lstm_cell_enc<<<1, 1024, 0, stream>>>(gates_e, b_ih_e, b_hh_e, eh, ec, eh_h);
    attn_kernel<<<1, 1024, 0, stream>>>(W_attn, b_attn, eh, dh, aw, din_h, t);
    gates_kernel<<<32, 256, 0, stream>>>(Wihd_h, 2 * H_DIM, din_h, 2 * H_DIM,
                                         Whhd_h, H_DIM, dh_h, H_DIM, gates_d);
    lstm_cell_dec<<<1, 1024, 0, stream>>>(gates_d, b_ih_d, b_hh_d, dh, dc, dh_h,
                                          din_h + H_DIM, W_out, b_out, out + t);
  }
}
